// GraphAttentionLayer_1065151890023
// MI455X (gfx1250) — matrix operand order repaired and checked by reading, not compiled, not run on hardware
//
#include <hip/hip_runtime.h>

typedef __attribute__((ext_vector_type(16))) __bf16 v16bf;
typedef __attribute__((ext_vector_type(8)))  float  v8f;

#define N_NODES 8192
#define F_IN    128
#define F_OUT   64
#define CHUNK   64                   // columns per fill (2 WMMA K-steps)
#define WAVES   8
#define NCHUNKS (N_NODES / CHUNK)    // 128
#define NITERS  (NCHUNKS / WAVES)    // 16 chunks per wave, uniform
#define MASKED  (-1.0e4f)            // exp(MASKED - anything_valid) == 0.0f

static __device__ __forceinline__ unsigned short f32_to_bf16_bits(float f) {
    union { float f; unsigned u; } v; v.f = f;
    unsigned r = (v.u + 0x7FFFu + ((v.u >> 16) & 1u)) >> 16;  // RNE
    return (unsigned short)r;
}

static __device__ __forceinline__ unsigned pack_bf16x2(float lo, float hi) {
    return (unsigned)f32_to_bf16_bits(lo) | ((unsigned)f32_to_bf16_bits(hi) << 16);
}

union FragU {                        // 32B: two b128 loads -> one WMMA operand
    struct { uint4 lo, hi; } q;
    v16bf v;
};

// ---------------- Kernel 1: h = X @ W ; store f32 h and bf16 h^T ------------
__global__ __launch_bounds__(256) void gemm_h_kernel(
    const float* __restrict__ inp, const float* __restrict__ W,
    float* __restrict__ hf, unsigned short* __restrict__ hbT)
{
    __shared__ float sW[F_IN * F_OUT];                 // 32 KB
    for (int t = threadIdx.x; t < F_IN * F_OUT; t += 256) sW[t] = W[t];
    __syncthreads();

    int idx = blockIdx.x * 256 + threadIdx.x;          // over N*F_OUT
    int i = idx >> 6, o = idx & 63;
    const float* ir = inp + (size_t)i * F_IN;
    float s = 0.f;
#pragma unroll 8
    for (int k = 0; k < F_IN; ++k) s = fmaf(ir[k], sW[k * F_OUT + o], s);
    hf[idx] = s;
    hbT[(size_t)o * N_NODES + i] = f32_to_bf16_bits(s);   // transposed bf16
}

// ---------------- Kernel 2: f1 = h@a1, f2 = h@a2 ----------------------------
__global__ __launch_bounds__(256) void fvec_kernel(
    const float* __restrict__ hf, const float* __restrict__ a,
    float* __restrict__ f1, float* __restrict__ f2)
{
    int i = blockIdx.x * 256 + threadIdx.x;            // one row per thread
    const float* hr = hf + (size_t)i * F_OUT;
    float s1 = 0.f, s2 = 0.f;
#pragma unroll 8
    for (int o = 0; o < F_OUT; ++o) {
        float hv = hr[o];
        s1 = fmaf(hv, a[o], s1);
        s2 = fmaf(hv, a[F_OUT + o], s2);
    }
    f1[i] = s1; f2[i] = s2;
}

// ---------------- Kernel 3: fused softmax-stats + P@h WMMA + ELU ------------
// Block = 16-row M-tile, 8 waves.
// Phase 1: each wave computes (rowmax, 1/rowsum) for 2 of the 16 rows
//          (coalesced adj reads; lines stay hot in L2 for phase 2).
// Phase 2: each wave owns disjoint 64-wide column chunks, builds a bf16
//          16x64 P tile in wave-private LDS (b64 adj loads, packed b32
//          stores), then runs 2 K-steps x 4 n-tiles of
//          v_wmma_f32_16x16x32_bf16. Cross-wave reduce in LDS, then ELU.
// Masked entries are handled by flooring e to MASKED before exp (selects on
// exp's input, not output) so the compiler cannot branch around the exp.
__global__ __launch_bounds__(256) void attn_kernel(
    const int* __restrict__ adj, const float* __restrict__ f1,
    const float* __restrict__ f2, const unsigned short* __restrict__ hbT,
    float* __restrict__ out)
{
    const int row0 = blockIdx.x * 16;
    const int lane = threadIdx.x & 31;
    const int wave = __builtin_amdgcn_readfirstlane(threadIdx.x >> 5);

    __shared__ __align__(16) unsigned short ptile[WAVES][16][CHUNK]; // 16 KB
    __shared__ float cbuf[WAVES][16][F_OUT];                         // 32 KB
    __shared__ float sf1[16], smax[16], sinv[16];

    // ---------------- Phase 1: row stats (max, 1/sum) ----------------
#pragma unroll 1
    for (int rr = 0; rr < 2; ++rr) {
        const int r   = 2 * wave + rr;
        const int row = row0 + r;
        const float f1i = f1[row];
        const int* ar = adj + (size_t)row * N_NODES;

        float m = MASKED, z = 0.f;
#pragma unroll 1
        for (int it = 0; it < N_NODES / 32; ++it) {    // uniform trip count
            const int j = lane + it * 32;
            __builtin_prefetch(ar + j + 256, 0, 1);    // global_prefetch_b8
            int av = ar[j];
            float e = f1i + f2[j];
            e = fmaxf(e, 0.2f * e);                    // leaky_relu(0.2)
            float e2 = (av > 0) ? e : MASKED;          // select BEFORE exp
            float mn = fmaxf(m, e2);
            if (mn > m) z *= __expf(m - mn);           // rare; kills pre-valid garbage
            z += __expf(e2 - mn);                      // masked -> +0 (underflow)
            m = mn;
        }
        // wave32 reduction of (m, z)
#pragma unroll
        for (int off = 16; off > 0; off >>= 1) {
            float mo = __shfl_xor(m, off, 32);
            float zo = __shfl_xor(z, off, 32);
            float mn = fmaxf(m, mo);
            z = z * __expf(m - mn) + zo * __expf(mo - mn);
            m = mn;
        }
        if (lane == 0) {
            sf1[r]  = f1i;
            smax[r] = m;
            sinv[r] = z > 0.f ? 1.0f / z : 0.f;
        }
    }
    __syncthreads();

    // hoist row stats into registers (asm memory clobbers below would
    // otherwise force per-chunk LDS reloads)
    float rf1[16], rmx[16], riz[16];
#pragma unroll
    for (int t = 0; t < 16; ++t) { rf1[t] = sf1[t]; rmx[t] = smax[t]; riz[t] = sinv[t]; }

    // ---------------- Phase 2: P @ h via WMMA ----------------
    v8f acc[4] = {};                                   // 16x64 f32 accum

    const int ahalf = lane >> 4;                       // A-frag lane half
    const int am    = lane & 15;                       // A-frag M row
    const int bn    = lane & 15;                       // B-frag N col
    const int bhalf = lane >> 4;                       // B-frag K half

#pragma unroll 1
    for (int itc = 0; itc < NITERS; ++itc) {           // uniform trip count
        const int j0 = (wave + itc * WAVES) * CHUNK;
        const float2 f2j = *(const float2*)(f2 + j0 + 2 * lane);

        // fill bf16 16x64 P tile: lane owns 2 adjacent columns
        // (b64 adj loads, packed b32 LDS stores, branchless)
#pragma unroll
        for (int t = 0; t < 16; ++t) {
            int2 av = *(const int2*)(adj + (size_t)(row0 + t) * N_NODES + j0 + 2 * lane);
            float e0 = rf1[t] + f2j.x;
            float e1 = rf1[t] + f2j.y;
            e0 = fmaxf(e0, 0.2f * e0);
            e1 = fmaxf(e1, 0.2f * e1);
            e0 = (av.x > 0) ? e0 : MASKED;             // select BEFORE exp
            e1 = (av.y > 0) ? e1 : MASKED;
            float p0 = __expf(e0 - rmx[t]) * riz[t];   // masked -> exact 0
            float p1 = __expf(e1 - rmx[t]) * riz[t];
            *(unsigned*)&ptile[wave][t][2 * lane] = pack_bf16x2(p0, p1);
        }
        // wave-private LDS tile: DS ops are in-order per wave, so a dscnt
        // wait + compiler barrier replaces a block-wide __syncthreads
        asm volatile("s_wait_dscnt 0x0" ::: "memory");

        // two K=32 steps over the 64-wide tile
#pragma unroll
        for (int kh = 0; kh < 2; ++kh) {
            const int jk = j0 + kh * 32;
            // A fragment, contiguous K order (same as B):
            // lanes 0-15: K 0..15 ; lanes 16-31: K 16..31
            const uint4* aprow = (const uint4*)&ptile[wave][am][kh * 32];
            FragU au;
            au.q.lo = aprow[2 * ahalf];
            au.q.hi = aprow[2 * ahalf + 1];

            // B fragments, same contiguous K order as A, from transposed h:
            // lane n reads 16 contiguous bf16 (K = bhalf*16 + 0..15) = 2x b128
#pragma unroll
            for (int nt = 0; nt < 4; ++nt) {
                const uint4* bp = (const uint4*)
                    (hbT + (size_t)(nt * 16 + bn) * N_NODES + jk + bhalf * 16);
                FragU bu;
                bu.q.lo = bp[0];
                bu.q.hi = bp[1];
                acc[nt] = __builtin_amdgcn_wmma_f32_16x16x32_bf16(
                    false, au.v, false, bu.v, (short)0, acc[nt], false, false);
            }
        }
        asm volatile("s_wait_dscnt 0x0" ::: "memory");  // WAR on ptile
    }

    // spill accumulators: C layout -> VGPR r: M = r + 8*(lane>=16), N = lane&15
    const int chalf = lane >> 4, cn = lane & 15;
#pragma unroll
    for (int nt = 0; nt < 4; ++nt)
#pragma unroll
        for (int r = 0; r < 8; ++r)
            cbuf[wave][r + 8 * chalf][nt * 16 + cn] = acc[nt][r];
    __syncthreads();

    // reduce 8 waves and apply ELU (uniform trip count: 4 reps of 256)
#pragma unroll 1
    for (int rep = 0; rep < (16 * F_OUT) / 256; ++rep) {
        int idx = threadIdx.x + rep * 256;
        int mm = idx >> 6, nn = idx & 63;
        float s = 0.f;
#pragma unroll
        for (int w = 0; w < WAVES; ++w) s += cbuf[w][mm][nn];
        out[(size_t)(row0 + mm) * F_OUT + nn] = s > 0.f ? s : (__expf(s) - 1.0f);
    }
}

// ---------------------------------------------------------------------------
extern "C" void kernel_launch(void* const* d_in, const int* in_sizes, int n_in,
                              void* d_out, int out_size, void* d_ws, size_t ws_size,
                              hipStream_t stream) {
    const float* inp = (const float*)d_in[0];   // 8192 x 128 f32
    const int*   adj = (const int*)  d_in[1];   // 8192 x 8192 i32
    const float* W   = (const float*)d_in[2];   // 128 x 64 f32
    const float* a   = (const float*)d_in[3];   // 128 x 1 f32
    float* out = (float*)d_out;                 // 8192 x 64 f32

    char* ws = (char*)d_ws;
    float*          hf  = (float*)ws;                                       // 2 MB
    unsigned short* hbT = (unsigned short*)(ws + (size_t)N_NODES*F_OUT*4);  // 1 MB (64 x 8192 bf16)
    float*          f1v = (float*)(ws + (size_t)N_NODES*F_OUT*6);
    float*          f2v = f1v + N_NODES;

    gemm_h_kernel<<<N_NODES * F_OUT / 256, 256, 0, stream>>>(inp, W, hf, hbT);
    fvec_kernel  <<<N_NODES / 256,        256, 0, stream>>>(hf, a, f1v, f2v);
    attn_kernel  <<<N_NODES / 16,         256, 0, stream>>>(adj, f1v, f2v, hbT, out);
}
